// LanguageModel_17806934410019
// MI455X (gfx1250) — matrix operand order repaired and checked by reading, not compiled, not run on hardware
//
#include <hip/hip_runtime.h>

// GPT forward pass on gfx1250 (MI455X): bf16 WMMA GEMMs + fused flash attention.
// All matrix math via v_wmma_f32_16x16x32_bf16, f32 accumulation.
// GEMM uses double-buffered LDS staging; A-tiles moved with
// global_load_async_to_lds_b128 (ASYNCcnt) when the builtin is available.

#define L_    6
#define H_    16
#define E_    1024
#define V_    32000
#define T_    1024
#define B_    4
#define DH_   64
#define ROWS_ (B_ * T_)     // 4096
#define FF_   (4 * E_)      // 4096

typedef __bf16 bf16;
typedef __attribute__((ext_vector_type(16))) __bf16 bf16x16;
typedef __attribute__((ext_vector_type(8)))  __bf16 bf16x8;
typedef __attribute__((ext_vector_type(8)))  float  v8f;
typedef int v4i __attribute__((vector_size(16)));   // matches async-LDS builtin param

#define AS1 __attribute__((address_space(1)))
#define AS3 __attribute__((address_space(3)))

#if defined(__has_builtin)
#if __has_builtin(__builtin_amdgcn_global_load_async_to_lds_b128) && \
    __has_builtin(__builtin_amdgcn_s_wait_asynccnt)
#define HAVE_ASYNC_LDS 1
#endif
#endif
#ifndef HAVE_ASYNC_LDS
#define HAVE_ASYNC_LDS 0
#endif

__device__ __forceinline__ v8f zero8() {
    v8f z;
#pragma unroll
    for (int i = 0; i < 8; ++i) z[i] = 0.0f;
    return z;
}

__device__ __forceinline__ bf16x16 combine8(bf16x8 lo, bf16x8 hi) {
    bf16x16 r;
#pragma unroll
    for (int i = 0; i < 8; ++i) { r[i] = lo[i]; r[i + 8] = hi[i]; }
    return r;
}

// A-operand fragment (16x32 bf16, MxK). Lane holds row M = lane&15.
// Element e: K = half*16 + e (run of 16 at row+half*16)
__device__ __forceinline__ bf16x16 frag_a(const bf16* row, int half) {
    bf16x8 lo = *(const bf16x8*)(row + half * 16);
    bf16x8 hi = *(const bf16x8*)(row + half * 16 + 8);
    return combine8(lo, hi);
}

// B-operand fragment (32x16 bf16, KxN) from K-contiguous row:
// lane holds col N = lane&15; element e: K = half*16 + e (run of 16)
__device__ __forceinline__ bf16x16 frag_b(const bf16* row, int half) {
    const bf16* p = row + half * 16;
    bf16x8 lo = *(const bf16x8*)p;
    bf16x8 hi = *(const bf16x8*)(p + 8);
    return combine8(lo, hi);
}

__device__ __forceinline__ v8f wmma_bf16(bf16x16 a, bf16x16 b, v8f c) {
    return __builtin_amdgcn_wmma_f32_16x16x32_bf16(
        false, a, false, b, (short)0, c, false, false);
}

// ---------------------------------------------------------------- embedding
__global__ __launch_bounds__(256) void embed_kernel(
    const int* __restrict__ idx, const float* __restrict__ tok,
    const float* __restrict__ pos, float* __restrict__ x) {
    int i = blockIdx.x * 256 + threadIdx.x;        // over ROWS_*E_
    int row = i >> 10;
    int e = i & (E_ - 1);
    int tk = idx[row];
    x[i] = tok[(size_t)tk * E_ + e] + pos[(size_t)(row & (T_ - 1)) * E_ + e];
}

// ---------------------------------------------------------------- layernorm
__global__ __launch_bounds__(256) void ln_kernel(
    const float* __restrict__ x, const float* __restrict__ w,
    const float* __restrict__ b, bf16* __restrict__ out) {
    __shared__ float s1[256];
    __shared__ float s2[256];
    const int tid = threadIdx.x;
    const int row = blockIdx.x;
    const float* xr = x + (size_t)row * E_;
    float sum = 0.f, sq = 0.f;
    for (int i = tid; i < E_; i += 256) { float v = xr[i]; sum += v; sq += v * v; }
    s1[tid] = sum; s2[tid] = sq;
    __syncthreads();
    for (int off = 128; off > 0; off >>= 1) {
        if (tid < off) { s1[tid] += s1[tid + off]; s2[tid] += s2[tid + off]; }
        __syncthreads();
    }
    float mu = s1[0] * (1.0f / E_);
    float var = s2[0] * (1.0f / E_) - mu * mu;
    float rs = rsqrtf(var + 1e-5f);
    bf16* o = out + (size_t)row * E_;
    for (int i = tid; i < E_; i += 256)
        o[i] = (bf16)((xr[i] - mu) * rs * w[i] + b[i]);
}

// ---------------------------------------------------------------- GEMM
// C[M,N] = A[M,K](bf16) * B[K,N](f32, converted) (+bias)(+resid)(relu?)
// 256 thr / 8 waves; tile 128x64; K-step 32; wave -> 32x32 (2x2 WMMA).
// Double-buffered LDS: stage tile k+1 while computing tile k.
// blayout: 0 = row-major B[k*N+n]; 1 = QKV head layout Wq[h][e][d], n=(h<<6)|d.
// Both reduce to a branch-free base+stride panel because n0 is 64-aligned.
__global__ __launch_bounds__(256) void gemm_kernel(
    const bf16* __restrict__ A, const float* __restrict__ B,
    const float* __restrict__ bias, const float* resid,
    float* Cf, bf16* Cb,
    int M, int N, int K, int blayout, int relu) {
    __shared__ __align__(16) bf16 As[2][128][40];
    __shared__ __align__(16) bf16 Bs[2][64][40];

    const int tid  = threadIdx.x;
    const int wave = tid >> 5;
    const int lane = tid & 31;
    const int half = lane >> 4;
    const int r16  = lane & 15;
    const int wm   = wave >> 1;            // 0..3 (M direction)
    const int wn   = wave & 1;             // 0..1 (N direction)
    const int m0   = blockIdx.y * 128;
    const int n0   = blockIdx.x * 64;
    (void)M;

    // branch-free B panel: element (k, n<64) at Bp[k*bstride + n]
    const float* Bp = B + (blayout ? (size_t)(n0 >> 6) * K * 64 : (size_t)n0);
    const int bstride = blayout ? 64 : N;

    // A staging: 512 chunks of 8 bf16 (16B); thread handles chunks tid, tid+256
    const int arow0 = tid >> 2;                 // chunk c=tid   -> row, kcol
    const int akc0  = (tid & 3) * 8;
    const int arow1 = (tid + 256) >> 2;         // chunk c=tid+256
    const int akc1  = ((tid + 256) & 3) * 8;

    // B staging: 512 float4 chunks; thread handles chunks tid, tid+256
    const int bk0 = tid >> 4;                   // 0..15
    const int bn0 = (tid & 15) * 4;
    const int bk1 = (tid + 256) >> 4;           // 16..31
    const int bn1 = bn0;

    auto stageA = [&](int bb, int kt) {
        const bf16* s0 = A + (size_t)(m0 + arow0) * K + kt + akc0;
        const bf16* s1 = A + (size_t)(m0 + arow1) * K + kt + akc1;
#if HAVE_ASYNC_LDS
        __builtin_amdgcn_global_load_async_to_lds_b128(
            (AS1 v4i*)(uintptr_t)s0, (AS3 v4i*)&As[bb][arow0][akc0], 0, 0);
        __builtin_amdgcn_global_load_async_to_lds_b128(
            (AS1 v4i*)(uintptr_t)s1, (AS3 v4i*)&As[bb][arow1][akc1], 0, 0);
#else
        bf16x8 v0 = *(const bf16x8*)s0;
        bf16x8 v1 = *(const bf16x8*)s1;
        *(bf16x8*)&As[bb][arow0][akc0] = v0;
        *(bf16x8*)&As[bb][arow1][akc1] = v1;
#endif
    };

    auto stageB = [&](int bb, int kt) {
        float4 va = *(const float4*)(Bp + (size_t)(kt + bk0) * bstride + bn0);
        float4 vb = *(const float4*)(Bp + (size_t)(kt + bk1) * bstride + bn1);
        Bs[bb][bn0 + 0][bk0] = (bf16)va.x;
        Bs[bb][bn0 + 1][bk0] = (bf16)va.y;
        Bs[bb][bn0 + 2][bk0] = (bf16)va.z;
        Bs[bb][bn0 + 3][bk0] = (bf16)va.w;
        Bs[bb][bn1 + 0][bk1] = (bf16)vb.x;
        Bs[bb][bn1 + 1][bk1] = (bf16)vb.y;
        Bs[bb][bn1 + 2][bk1] = (bf16)vb.z;
        Bs[bb][bn1 + 3][bk1] = (bf16)vb.w;
    };

    v8f acc[2][2];
    acc[0][0] = zero8(); acc[0][1] = zero8();
    acc[1][0] = zero8(); acc[1][1] = zero8();

    stageA(0, 0);
    stageB(0, 0);

    int buf = 0;
    for (int kt = 0; kt < K; kt += 32, buf ^= 1) {
#if HAVE_ASYNC_LDS
        __builtin_amdgcn_s_wait_asynccnt(0);
#endif
        __syncthreads();
        if (kt + 32 < K) {
            stageA(buf ^ 1, kt + 32);
            stageB(buf ^ 1, kt + 32);
        }
        bf16x16 a0 = frag_a(&As[buf][wm * 32 + r16][0],      half);
        bf16x16 a1 = frag_a(&As[buf][wm * 32 + 16 + r16][0], half);
        bf16x16 b0 = frag_b(&Bs[buf][wn * 32 + r16][0],      half);
        bf16x16 b1 = frag_b(&Bs[buf][wn * 32 + 16 + r16][0], half);
        acc[0][0] = wmma_bf16(a0, b0, acc[0][0]);
        acc[0][1] = wmma_bf16(a0, b1, acc[0][1]);
        acc[1][0] = wmma_bf16(a1, b0, acc[1][0]);
        acc[1][1] = wmma_bf16(a1, b1, acc[1][1]);
    }

    // epilogue: D layout -> row = m0+wm*32+mi*16+half*8+r, col = n0+wn*32+ni*16+r16
#pragma unroll
    for (int mi = 0; mi < 2; ++mi)
#pragma unroll
        for (int ni = 0; ni < 2; ++ni)
#pragma unroll
            for (int r = 0; r < 8; ++r) {
                int row = m0 + wm * 32 + mi * 16 + half * 8 + r;
                int col = n0 + wn * 32 + ni * 16 + r16;
                float vv = acc[mi][ni][r];
                if (bias)  vv += bias[col];
                if (resid) vv += resid[(size_t)row * N + col];
                if (relu)  vv = fmaxf(vv, 0.0f);
                if (Cf) Cf[(size_t)row * N + col] = vv;
                if (Cb) Cb[(size_t)row * N + col] = (bf16)vv;
            }
}

// ---------------------------------------------------------------- attention
// Flash-style. One wave per 16-row Q tile of one (b,h). q/k/v/o: [B*T, E] bf16
// with head h occupying columns h*64 .. h*64+63.
__global__ __launch_bounds__(128) void attn_kernel(
    const bf16* __restrict__ qm, const bf16* __restrict__ km,
    const bf16* __restrict__ vm, bf16* __restrict__ om) {
    __shared__ __align__(16) bf16 Plds[4][16][40];

    const int tid  = threadIdx.x;
    const int wave = tid >> 5;
    const int lane = tid & 31;
    const int half = lane >> 4;
    const int r16  = lane & 15;

    const int g    = blockIdx.x * 4 + wave;    // 0..4095 tiles
    const int qt   = g & 63;
    const int h    = (g >> 6) & 15;
    const int bidx = g >> 10;
    const int t0   = qt * 16;
    const int bT   = bidx * T_;
    const int hcol = h * DH_;
    const float SCALE = 0.125f;                // Dh^-0.5

    const bf16* qrow = qm + (size_t)(bT + t0 + r16) * E_ + hcol;
    bf16x16 aQ0 = frag_a(qrow, half);
    bf16x16 aQ1 = frag_a(qrow + 32, half);

    float mrow[8], lrow[8];
#pragma unroll
    for (int r = 0; r < 8; ++r) { mrow[r] = -1e30f; lrow[r] = 0.0f; }
    v8f oacc[4];
#pragma unroll
    for (int ni = 0; ni < 4; ++ni) oacc[ni] = zero8();

    for (int s0 = 0; s0 < t0 + 16; s0 += 32) {
        const bf16* krow0 = km + (size_t)(bT + s0 + r16) * E_ + hcol;
        const bf16* krow1 = km + (size_t)(bT + s0 + 16 + r16) * E_ + hcol;
        v8f sc0 = zero8(), sc1 = zero8();
        sc0 = wmma_bf16(aQ0, frag_b(krow0, half),      sc0);
        sc0 = wmma_bf16(aQ1, frag_b(krow0 + 32, half), sc0);
        sc1 = wmma_bf16(aQ0, frag_b(krow1, half),      sc1);
        sc1 = wmma_bf16(aQ1, frag_b(krow1 + 32, half), sc1);

        // online softmax per row (row r of this half lives in one 16-lane group)
#pragma unroll
        for (int r = 0; r < 8; ++r) {
            int t = t0 + half * 8 + r;
            float va = (s0 + r16 <= t)      ? sc0[r] * SCALE : -1e30f;
            float vb = (s0 + 16 + r16 <= t) ? sc1[r] * SCALE : -1e30f;
            float mx = fmaxf(va, vb);
            mx = fmaxf(mx, __shfl_xor(mx, 1, 16));
            mx = fmaxf(mx, __shfl_xor(mx, 2, 16));
            mx = fmaxf(mx, __shfl_xor(mx, 4, 16));
            mx = fmaxf(mx, __shfl_xor(mx, 8, 16));
            float mnew = fmaxf(mrow[r], mx);
            float al   = __expf(mrow[r] - mnew);
            mrow[r] = mnew;
            float pa = __expf(va - mnew);
            float pb = __expf(vb - mnew);
            float rs = pa + pb;
            rs += __shfl_xor(rs, 1, 16);
            rs += __shfl_xor(rs, 2, 16);
            rs += __shfl_xor(rs, 4, 16);
            rs += __shfl_xor(rs, 8, 16);
            lrow[r] = lrow[r] * al + rs;
            oacc[0][r] *= al; oacc[1][r] *= al;
            oacc[2][r] *= al; oacc[3][r] *= al;
            Plds[wave][half * 8 + r][r16]      = (bf16)pa;
            Plds[wave][half * 8 + r][16 + r16] = (bf16)pb;
        }

        // O += P (16x32) * V (32x64)
        bf16x16 aP = frag_a(&Plds[wave][r16][0], half);
        const bf16* vbase = vm + (size_t)(bT + s0 + half * 16) * E_ + hcol;
#pragma unroll
        for (int ni = 0; ni < 4; ++ni) {
            bf16x16 bV;
            const bf16* vp = vbase + ni * 16 + r16;
#pragma unroll
            for (int e = 0; e < 16; ++e) bV[e] = vp[(size_t)e * E_];
            oacc[ni] = wmma_bf16(aP, bV, oacc[ni]);
        }
    }

#pragma unroll
    for (int ni = 0; ni < 4; ++ni)
#pragma unroll
        for (int r = 0; r < 8; ++r) {
            int t = t0 + half * 8 + r;
            om[(size_t)(bT + t) * E_ + hcol + ni * 16 + r16] =
                (bf16)(oacc[ni][r] / lrow[r]);
        }
}

// ---------------------------------------------------------------- loss
__global__ void init_loss_kernel(float* loss) { *loss = 0.0f; }

__global__ __launch_bounds__(256) void loss_kernel(
    const float* __restrict__ logits, const int* __restrict__ targets,
    float* __restrict__ loss) {
    __shared__ float red[256];
    const int tid = threadIdx.x;
    const int row = blockIdx.x;
    const float* lr = logits + (size_t)row * V_;
    float mx = -1e30f;
    for (int i = tid; i < V_; i += 256) mx = fmaxf(mx, lr[i]);
    red[tid] = mx;
    __syncthreads();
    for (int off = 128; off > 0; off >>= 1) {
        if (tid < off) red[tid] = fmaxf(red[tid], red[tid + off]);
        __syncthreads();
    }
    mx = red[0];
    __syncthreads();
    float se = 0.f;
    for (int i = tid; i < V_; i += 256) se += __expf(lr[i] - mx);
    red[tid] = se;
    __syncthreads();
    for (int off = 128; off > 0; off >>= 1) {
        if (tid < off) red[tid] += red[tid + off];
        __syncthreads();
    }
    if (tid == 0) {
        int t = targets[row];
        float lp = lr[t] - mx - __logf(red[0]);
        atomicAdd(loss, -lp * (1.0f / ROWS_));
    }
}

// ---------------------------------------------------------------- launch
extern "C" void kernel_launch(void* const* d_in, const int* in_sizes, int n_in,
                              void* d_out, int out_size, void* d_ws, size_t ws_size,
                              hipStream_t stream) {
    (void)in_sizes; (void)n_in; (void)out_size; (void)ws_size;

    const int*   idx = (const int*)d_in[0];
    const int*   tgt = (const int*)d_in[1];
    const float* tok = (const float*)d_in[2];
    const float* pos = (const float*)d_in[3];
    const float* Wq  = (const float*)d_in[4];
    const float* Wk  = (const float*)d_in[5];
    const float* Wv  = (const float*)d_in[6];
    const float* Wp  = (const float*)d_in[7];
    const float* bp  = (const float*)d_in[8];
    const float* W1  = (const float*)d_in[9];
    const float* b1  = (const float*)d_in[10];
    const float* W2  = (const float*)d_in[11];
    const float* b2  = (const float*)d_in[12];
    const float* l1w = (const float*)d_in[13];
    const float* l1b = (const float*)d_in[14];
    const float* l2w = (const float*)d_in[15];
    const float* l2b = (const float*)d_in[16];
    const float* lfw = (const float*)d_in[17];
    const float* lfb = (const float*)d_in[18];
    const float* Wlm = (const float*)d_in[19];
    const float* blm = (const float*)d_in[20];

    // workspace bump allocator (256B aligned)
    char* wsp = (char*)d_ws;
    auto balloc = [&](size_t bytes) {
        void* r = (void*)wsp;
        wsp += (bytes + 255) & ~(size_t)255;
        return r;
    };
    float* x  = (float*)balloc((size_t)ROWS_ * E_ * 4);
    bf16*  h  = (bf16*)balloc((size_t)ROWS_ * E_ * 2);
    bf16*  qb = (bf16*)balloc((size_t)ROWS_ * E_ * 2);
    bf16*  kb = (bf16*)balloc((size_t)ROWS_ * E_ * 2);
    bf16*  vb = (bf16*)balloc((size_t)ROWS_ * E_ * 2);
    bf16*  ob = (bf16*)balloc((size_t)ROWS_ * E_ * 2);
    bf16*  ub = (bf16*)balloc((size_t)ROWS_ * FF_ * 2);

    embed_kernel<<<ROWS_ * E_ / 256, 256, 0, stream>>>(idx, tok, pos, x);

    const dim3 gE(E_ / 64, ROWS_ / 128);       // N=1024 GEMMs
    const dim3 gF(FF_ / 64, ROWS_ / 128);      // N=4096 GEMM

    for (int l = 0; l < L_; ++l) {
        const size_t wqo = (size_t)l * E_ * E_;
        ln_kernel<<<ROWS_, 256, 0, stream>>>(x, l1w + l * E_, l1b + l * E_, h);
        gemm_kernel<<<gE, 256, 0, stream>>>(h, Wq + wqo, nullptr, nullptr,
                                            nullptr, qb, ROWS_, E_, E_, 1, 0);
        gemm_kernel<<<gE, 256, 0, stream>>>(h, Wk + wqo, nullptr, nullptr,
                                            nullptr, kb, ROWS_, E_, E_, 1, 0);
        gemm_kernel<<<gE, 256, 0, stream>>>(h, Wv + wqo, nullptr, nullptr,
                                            nullptr, vb, ROWS_, E_, E_, 1, 0);
        attn_kernel<<<(B_ * H_ * (T_ / 16)) / 4, 128, 0, stream>>>(qb, kb, vb, ob);
        gemm_kernel<<<gE, 256, 0, stream>>>(ob, Wp + wqo, bp + l * E_, x,
                                            x, nullptr, ROWS_, E_, E_, 0, 0);
        ln_kernel<<<ROWS_, 256, 0, stream>>>(x, l2w + l * E_, l2b + l * E_, h);
        gemm_kernel<<<gF, 256, 0, stream>>>(h, W1 + (size_t)l * E_ * FF_,
                                            b1 + (size_t)l * FF_, nullptr,
                                            nullptr, ub, ROWS_, FF_, E_, 0, 1);
        gemm_kernel<<<gE, 256, 0, stream>>>(ub, W2 + (size_t)l * FF_ * E_,
                                            b2 + l * E_, x,
                                            x, nullptr, ROWS_, E_, FF_, 0, 0);
    }

    ln_kernel<<<ROWS_, 256, 0, stream>>>(x, lfw, lfb, h);

    float* logits = (float*)d_out;
    float* loss   = logits + (size_t)ROWS_ * V_;
    const dim3 gL(V_ / 64, ROWS_ / 128);       // 32000/64 = 500
    gemm_kernel<<<gL, 256, 0, stream>>>(h, Wlm, blm, nullptr,
                                        logits, nullptr, ROWS_, V_, E_, 0, 0);
    init_loss_kernel<<<1, 1, 0, stream>>>(loss);
    loss_kernel<<<ROWS_, 256, 0, stream>>>(logits, tgt, loss);
}
